// Deductron_52905407152422
// MI455X (gfx1250) — matrix operand order repaired and checked by reading, not compiled, not run on hardware
//
#include <hip/hip_runtime.h>
#include <hip/hip_bf16.h>

// MI455X (gfx1250) implementation of the "Deductron":
//   h = sigmoid(X@W1+B1); a = h_left*h_right; b = 1-h_left;
//   u_t = a_{t-1} u_{t-1} + b_{t-1} (scan); out = z@W2+B2.
//
// Strategy (memory-bound, 23.3 TB/s HBM):
//   Pass 1 (compose): per 512-row chunk, recompute h tiles with
//     v_wmma_f32_16x16x32_bf16, fold a,b into per-chunk affine (A,B).
//   Spine: 1 block scans the 1024 chunk aggregates -> u_start per chunk.
//   Pass 2 (apply): recompute h tiles, run the local scan in f32
//     (1 lane = 1 channel), write z to LDS, GEMM2 via WMMA, store out.
// X is read twice, out written once (~335 MB total HBM traffic); a,b,z
// never touch HBM. Workspace use: 3 * nchunk * 64 floats (~768 KB).
//
// CDNA5-specific data movement: X tiles are streamed into a double-buffered
// LDS tile with GLOBAL_LOAD_ASYNC_TO_LDS_B128 (ASYNCcnt), issuing tile it+1
// before waiting on tile it, so HBM latency overlaps WMMA+scan compute.
// Sigmoid uses v_exp_f32 + raw v_rcp_f32 (no IEEE divide expansion).

typedef __bf16 v16bf __attribute__((ext_vector_type(16)));
typedef float  v8f   __attribute__((ext_vector_type(8)));
typedef int    v4i   __attribute__((ext_vector_type(4)));

typedef __attribute__((address_space(1))) v4i* gptr_b128;  // global int4*
typedef __attribute__((address_space(3))) v4i* lptr_b128;  // LDS int4*

#define LDSS 68   // padded LDS row stride (floats) to break bank conflicts
#define CHUNK 512

static __device__ inline v8f wmma_bf16(v16bf a, v16bf b, v8f c) {
  return __builtin_amdgcn_wmma_f32_16x16x32_bf16(
      /*neg_a=*/false, a, /*neg_b=*/false, b,
      /*c_mod=*/(short)0, c, /*reuse_a=*/false, /*reuse_b=*/false);
}

// Async 16-byte copy global -> LDS (no VGPR staging, tracked by ASYNCcnt).
static __device__ inline void async_copy_b128(const float* gsrc, float* ldst) {
#if defined(__has_builtin) && __has_builtin(__builtin_amdgcn_global_load_async_to_lds_b128)
  __builtin_amdgcn_global_load_async_to_lds_b128(
      (gptr_b128)gsrc, (lptr_b128)ldst, 0, 0);
#else
  unsigned lofs = (unsigned)(uintptr_t)(lptr_b128)ldst;
  asm volatile("global_load_async_to_lds_b128 %0, %1, off"
               :: "v"(lofs), "v"(gsrc) : "memory");
#endif
}

// Wait until at most 1 async copy is outstanding (the just-issued next tile).
static __device__ inline void wait_asynccnt_le1() {
#if defined(__has_builtin) && __has_builtin(__builtin_amdgcn_s_wait_asynccnt)
  __builtin_amdgcn_s_wait_asynccnt(1);
#else
  asm volatile("s_wait_asynccnt 0x1" ::: "memory");
#endif
}

// A fragment (16x32 bf16) for M=16 rows, K in [kbase, kbase+32), from f32 LDS.
// Lane L: row M = L&15; element e -> K = e + 16*(L>=16).
static __device__ inline v16bf load_a_frag(const float* lds, int kbase, int lane) {
  int m    = lane & 15;
  int koff = (lane >> 4) << 4;
  v16bf f;
#pragma unroll
  for (int e = 0; e < 16; ++e) {
    int k = e + koff;
    f[e] = (__bf16)lds[m * LDSS + kbase + k];
  }
  return f;
}

// B fragment (32x16 bf16): rows K in [kbase,kbase+32), cols [nbase,nbase+16)
// of row-major f32 W with leading dim ldw. Lane L: col N = L&15;
// element e -> K = kbase + e + 16*(L>=16).
static __device__ inline v16bf load_b_frag(const float* __restrict__ W, int ldw,
                                           int kbase, int nbase, int lane) {
  int n    = nbase + (lane & 15);
  int koff = kbase + ((lane >> 4) << 4);
  v16bf f;
#pragma unroll
  for (int e = 0; e < 16; ++e) f[e] = (__bf16)W[(koff + e) * ldw + n];
  return f;
}

// Fast sigmoid: v_exp_f32 (via __expf) + v_add + raw v_rcp_f32.
static __device__ inline float sigmoidf_fast(float x) {
  return __builtin_amdgcn_rcpf(1.0f + __expf(-x));
}

// ---------------------------------------------------------------------------
// Pass 1: per-chunk affine composition (A,B) per channel.
// Block = 256 threads (8 waves); wave w owns h column tile [16w,16w+16).
// ---------------------------------------------------------------------------
__global__ __launch_bounds__(256) void ded_compose(
    const float* __restrict__ X, const float* __restrict__ W1,
    const float* __restrict__ B1, float* __restrict__ chunkA,
    float* __restrict__ chunkB, int T) {
  __shared__ __align__(16) float lx[2][16 * LDSS];  // double-buffered X tile
  __shared__ float la[16 * LDSS];   // h left  (cols 0..63)
  __shared__ float lb[16 * LDSS];   // h right (cols 64..127)

  const int tid = threadIdx.x, lane = tid & 31, wave = tid >> 5;
  const int nbase = wave * 16;
  const int rr = tid >> 4, cc = (tid & 15) << 2;   // this thread's 16B of X tile

  // W1 B-fragments: constant across the whole chunk, load once per wave.
  const v16bf wb0 = load_b_frag(W1, 128, 0, nbase, lane);
  const v16bf wb1 = load_b_frag(W1, 128, 32, nbase, lane);
  const float bias = B1[nbase + (lane & 15)];

  float Ac = 1.0f, Bc = 0.0f;               // running affine map (lane=channel)
  const int r0 = blockIdx.x * CHUNK;
  const int avail = (T - r0 < CHUNK) ? (T - r0) : CHUNK;
  const int ntile = avail >> 4;
  if (ntile <= 0) return;

  // Prologue: kick off tile 0 into buffer 0.
  async_copy_b128(X + (size_t)(r0 + rr) * 64 + cc, &lx[0][rr * LDSS + cc]);

  for (int it = 0; it < ntile; ++it) {
    const int rows = r0 + (it << 4);
    const int cur = it & 1;
    __syncthreads();   // everyone done reading lx[cur^1] / la / lb

    // Issue next tile (or re-load current on the last iteration) -> other buf.
    const int nb = (it + 1 < ntile) ? rows + 16 : rows;
    async_copy_b128(X + (size_t)(nb + rr) * 64 + cc,
                    &lx[cur ^ 1][rr * LDSS + cc]);

    wait_asynccnt_le1();   // in-order completion: current tile is resident
    __syncthreads();       // ...across all 8 waves

    v16bf a0 = load_a_frag(lx[cur], 0, lane);
    v16bf a1 = load_a_frag(lx[cur], 32, lane);
    v8f c;
#pragma unroll
    for (int r = 0; r < 8; ++r) c[r] = bias;
    c = wmma_bf16(a0, wb0, c);              // K = 0..31
    c = wmma_bf16(a1, wb1, c);              // K = 32..63

    // sigmoid, scatter into left/right LDS tiles (C layout: lane=col, vgpr=row)
    float* dst = (wave < 4) ? la : lb;
    const int col = ((wave < 4) ? nbase : nbase - 64) + (lane & 15);
    const int rowoff = (lane >> 4) << 3;
#pragma unroll
    for (int r = 0; r < 8; ++r)
      dst[(r + rowoff) * LDSS + col] = sigmoidf_fast(c[r]);
    __syncthreads();

    // Sequential (in-order) composition over the 16 rows: 1 lane = 1 channel.
    if (tid < 64) {
#pragma unroll
      for (int t = 0; t < 16; ++t) {
        float l = la[t * LDSS + tid];
        float rv = lb[t * LDSS + tid];
        float av = l * rv, bv = 1.0f - l;
        Ac = av * Ac;
        Bc = av * Bc + bv;
      }
    }
  }
  if (tid < 64) {
    chunkA[blockIdx.x * 64 + tid] = Ac;
    chunkB[blockIdx.x * 64 + tid] = Bc;
  }
}

// ---------------------------------------------------------------------------
// Spine: exclusive scan over chunk aggregates. 1 block, 64 lanes = channels.
// Loads are iteration-independent (pipelined); dep chain is FMAs only.
// ---------------------------------------------------------------------------
__global__ void ded_spine(const float* __restrict__ chunkA,
                          const float* __restrict__ chunkB,
                          float* __restrict__ ustart, int nchunk) {
  const int ch = threadIdx.x;
  float u = 0.0f;
#pragma unroll 4
  for (int c = 0; c < nchunk; ++c) {
    ustart[c * 64 + ch] = u;
    u = chunkA[c * 64 + ch] * u + chunkB[c * 64 + ch];
  }
}

// ---------------------------------------------------------------------------
// Pass 2: apply scan locally, z -> GEMM2 via WMMA straight from LDS.
// ---------------------------------------------------------------------------
__global__ __launch_bounds__(256) void ded_apply(
    const float* __restrict__ X, const float* __restrict__ W1,
    const float* __restrict__ B1, const float* __restrict__ W2,
    const float* __restrict__ B2, const float* __restrict__ ustart,
    float* __restrict__ out, int T) {
  __shared__ __align__(16) float lx[2][16 * LDSS];
  __shared__ float la[16 * LDSS];
  __shared__ float lb[16 * LDSS];
  __shared__ float lz[16 * LDSS];

  const int tid = threadIdx.x, lane = tid & 31, wave = tid >> 5;
  const int nbase = wave * 16;
  const int rr = tid >> 4, cc = (tid & 15) << 2;

  const v16bf wb0 = load_b_frag(W1, 128, 0, nbase, lane);
  const v16bf wb1 = load_b_frag(W1, 128, 32, nbase, lane);
  const float bias = B1[nbase + (lane & 15)];

  // W2 fragments for the two 16-col output tiles (waves 0 and 1 only).
  v16bf w2b0{}, w2b1{};
  float bias2 = 0.0f;
  if (wave < 2) {
    w2b0 = load_b_frag(W2, 32, 0, wave * 16, lane);
    w2b1 = load_b_frag(W2, 32, 32, wave * 16, lane);
    bias2 = B2[wave * 16 + (lane & 15)];
  }

  float u = 0.0f;                 // z at the top of this chunk (lane=channel)
  if (tid < 64) u = ustart[blockIdx.x * 64 + tid];
  const int r0 = blockIdx.x * CHUNK;
  const int avail = (T - r0 < CHUNK) ? (T - r0) : CHUNK;
  const int ntile = avail >> 4;
  if (ntile <= 0) return;

  async_copy_b128(X + (size_t)(r0 + rr) * 64 + cc, &lx[0][rr * LDSS + cc]);

  for (int it = 0; it < ntile; ++it) {
    const int rows = r0 + (it << 4);
    const int cur = it & 1;
    __syncthreads();

    const int nb = (it + 1 < ntile) ? rows + 16 : rows;
    async_copy_b128(X + (size_t)(nb + rr) * 64 + cc,
                    &lx[cur ^ 1][rr * LDSS + cc]);

    wait_asynccnt_le1();
    __syncthreads();

    // GEMM1 tile + sigmoid (same as pass 1)
    v16bf a0 = load_a_frag(lx[cur], 0, lane);
    v16bf a1 = load_a_frag(lx[cur], 32, lane);
    v8f c;
#pragma unroll
    for (int r = 0; r < 8; ++r) c[r] = bias;
    c = wmma_bf16(a0, wb0, c);
    c = wmma_bf16(a1, wb1, c);
    float* dst = (wave < 4) ? la : lb;
    const int col = ((wave < 4) ? nbase : nbase - 64) + (lane & 15);
    const int rowoff = (lane >> 4) << 3;
#pragma unroll
    for (int r = 0; r < 8; ++r)
      dst[(r + rowoff) * LDSS + col] = sigmoidf_fast(c[r]);
    __syncthreads();

    // Local scan: z[rows+0] = u; z[rows+j] uses a,b rows rows..rows+j-1.
    if (tid < 64) {
      lz[0 * LDSS + tid] = u;
#pragma unroll
      for (int t = 0; t < 15; ++t) {
        float l = la[t * LDSS + tid];
        float rv = lb[t * LDSS + tid];
        u = (l * rv) * u + (1.0f - l);
        lz[(t + 1) * LDSS + tid] = u;
      }
      {   // carry into next tile (not stored)
        float l = la[15 * LDSS + tid];
        float rv = lb[15 * LDSS + tid];
        u = (l * rv) * u + (1.0f - l);
      }
    }
    __syncthreads();

    // GEMM2: (16x64 z) @ (64x32 W2); wave-uniform branch keeps EXEC all-ones.
    if (wave < 2) {
      v16bf za0 = load_a_frag(lz, 0, lane);
      v16bf za1 = load_a_frag(lz, 32, lane);
      v8f c2;
#pragma unroll
      for (int r = 0; r < 8; ++r) c2[r] = bias2;
      c2 = wmma_bf16(za0, w2b0, c2);
      c2 = wmma_bf16(za1, w2b1, c2);
      const int ocol = wave * 16 + (lane & 15);
#pragma unroll
      for (int r = 0; r < 8; ++r)
        out[(size_t)(rows + r + rowoff) * 32 + ocol] = c2[r];
    }
  }
}

extern "C" void kernel_launch(void* const* d_in, const int* in_sizes, int n_in,
                              void* d_out, int out_size, void* d_ws, size_t ws_size,
                              hipStream_t stream) {
  const float* X  = (const float*)d_in[0];   // (T, 64)
  const float* W1 = (const float*)d_in[1];   // (64, 128)
  const float* B1 = (const float*)d_in[2];   // (1, 128)
  const float* W2 = (const float*)d_in[3];   // (64, 32)
  const float* B2 = (const float*)d_in[4];   // (1, 32)
  float* out = (float*)d_out;                // (T, 32)

  const int T = in_sizes[0] / 64;            // 524288
  const int nchunk = (T + CHUNK - 1) / CHUNK;

  float* chunkA = (float*)d_ws;
  float* chunkB = chunkA + (size_t)nchunk * 64;
  float* ustart = chunkB + (size_t)nchunk * 64;

  ded_compose<<<nchunk, 256, 0, stream>>>(X, W1, B1, chunkA, chunkB, T);
  ded_spine<<<1, 64, 0, stream>>>(chunkA, chunkB, ustart, nchunk);
  ded_apply<<<nchunk, 256, 0, stream>>>(X, W1, B1, W2, B2, ustart, out, T);
}
